// NetVLAD_63977832841863
// MI455X (gfx1250) — compile-verified, matrix operand order checked by reading, not run on hardware
//
#include <hip/hip_runtime.h>
#include <hip/hip_bf16.h>
#include <stdint.h>

// ---------------- CDNA5 WMMA types ----------------
typedef __attribute__((ext_vector_type(16))) __bf16 v16bf;
typedef __attribute__((ext_vector_type(8)))  float  v8f;

#define B_ 32
#define M_ 2048
#define D_ 1024
#define K_ 64

// ---- bf16 pack: v_cvt_pk_bf16_f32 via inline asm (1 instr, RNE) ----
// clang-22 for gfx1250 does not declare __builtin_amdgcn_cvt_pk_bf16_f32,
// but the ISA has the instruction; emit it directly.
__device__ __forceinline__ unsigned pk2(float a, float b) {
    unsigned d;
    asm("v_cvt_pk_bf16_f32 %0, %1, %2" : "=v"(d) : "v"(a), "v"(b));
    return d;
}
__device__ __forceinline__ unsigned short f2bf(float x) {
    return (unsigned short)(pk2(x, 0.f) & 0xFFFFu);
}

// ---- cross-lane butterflies via ds_swizzle_b32 immediate patterns ----
// group-of-32 encoding: offset[14:10]=xor_mask, [9:5]=or_mask, [4:0]=and_mask
template <int IMM>
__device__ __forceinline__ float swzf(float v) {
    union { float f; int i; } u; u.f = v;
    u.i = __builtin_amdgcn_ds_swizzle(u.i, IMM);
    return u.f;
}
__device__ __forceinline__ float wave_red_add(float v) {   // full 32-lane sum
    v += swzf<0x041f>(v); v += swzf<0x081f>(v); v += swzf<0x101f>(v);
    v += swzf<0x201f>(v); v += swzf<0x401f>(v);
    return v;
}
__device__ __forceinline__ float half_red_add(float v) {   // 16-lane-half sum
    v += swzf<0x041f>(v); v += swzf<0x081f>(v);
    v += swzf<0x101f>(v); v += swzf<0x201f>(v);
    return v;
}
__device__ __forceinline__ float half_red_max(float v) {   // 16-lane-half max
    v = fmaxf(v, swzf<0x041f>(v)); v = fmaxf(v, swzf<0x081f>(v));
    v = fmaxf(v, swzf<0x101f>(v)); v = fmaxf(v, swzf<0x201f>(v));
    return v;
}

// ---------------- zero atomic accumulators ----------------
__global__ void k_zero(float* __restrict__ s, float* __restrict__ ssb) {
    int i = blockIdx.x * 256 + threadIdx.x;
    if (i < B_ * K_) s[i] = 0.f;
    if (i < B_)      ssb[i] = 0.f;
}

// ---------------- per-row 1/||x|| ----------------
__global__ void k_rownorm(const float* __restrict__ x, float* __restrict__ invn) {
    int lane = threadIdx.x & 31, w = threadIdx.x >> 5;
    int row = blockIdx.x * 8 + w;                       // B_*M_ rows
    const float* xr = x + (size_t)row * D_;
    float ss = 0.f;
    #pragma unroll
    for (int i = 0; i < 8; ++i) {
        float4 v = *(const float4*)(xr + i * 128 + lane * 4);
        ss += v.x * v.x + v.y * v.y + v.z * v.z + v.w * v.w;
    }
    ss = wave_red_add(ss);
    if (lane == 0) invn[row] = 1.f / fmaxf(sqrtf(ss), 1e-12f);
}

// ---------------- W -> bf16 (no transpose: WMMA B-operand columns == W rows) ----------------
__global__ void k_wconv(const float* __restrict__ W, unsigned short* __restrict__ Wbf) {
    int i = blockIdx.x * 256 + threadIdx.x;             // K_*D_ = 65536
    Wbf[i] = f2bf(W[i]);
}

// ---------------- x * invnorm -> bf16, transposed: xn_t[b][d][m] ----------------
__global__ void k_transpose(const float* __restrict__ x, const float* __restrict__ invn,
                            unsigned short* __restrict__ xnt) {
    __shared__ unsigned short tile[32][33];
    int b = blockIdx.z, m0 = blockIdx.y * 32, d0 = blockIdx.x * 32;
    int tx = threadIdx.x, ty = threadIdx.y;
    #pragma unroll
    for (int i = 0; i < 4; ++i) {
        int ml = ty + 8 * i;
        int row = b * M_ + m0 + ml;
        float v = x[(size_t)row * D_ + d0 + tx] * invn[row];
        tile[ml][tx] = f2bf(v);
    }
    __syncthreads();
    #pragma unroll
    for (int i = 0; i < 4; ++i) {
        int dl = ty + 8 * i;
        xnt[((size_t)b * D_ + d0 + dl) * M_ + m0 + tx] = tile[tx][dl];
    }
}

// ---------------- GEMM1 + softmax: a_t[b][k][m] bf16, colsums s[b][k] ----------------
// One wave: 16 rows (m) x 64 cols (k). A-frag converted from fp32 x on the fly.
__global__ void k_logits_softmax(const float* __restrict__ x, const float* __restrict__ invn,
                                 const unsigned short* __restrict__ Wbf,
                                 unsigned short* __restrict__ at, float* __restrict__ s) {
    int lane = threadIdx.x & 31, w = threadIdx.x >> 5;
    int gw = blockIdx.x * 4 + w;                        // 4096 waves
    int b = gw >> 7, m0 = (gw & 127) * 16;
    int lrow = lane & 15, hf = lane >> 4;
    int row = b * M_ + m0 + lrow;
    const float* xr = x + (size_t)row * D_;
    float inv_m = invn[row];

    v8f c[4] = {};
    union AV { v16bf v; unsigned u[8]; } A;
    union BV { v16bf v; uint4 q[2]; } Bv;

    for (int d0 = 0; d0 < D_; d0 += 32) {
        int co = d0 + hf * 8;                           // K-chunk pattern per ISA A-layout
        float4 f0 = *(const float4*)(xr + co);
        float4 f1 = *(const float4*)(xr + co + 4);
        float4 f2 = *(const float4*)(xr + co + 16);
        float4 f3 = *(const float4*)(xr + co + 20);
        A.u[0] = pk2(f0.x * inv_m, f0.y * inv_m); A.u[1] = pk2(f0.z * inv_m, f0.w * inv_m);
        A.u[2] = pk2(f1.x * inv_m, f1.y * inv_m); A.u[3] = pk2(f1.z * inv_m, f1.w * inv_m);
        A.u[4] = pk2(f2.x * inv_m, f2.y * inv_m); A.u[5] = pk2(f2.z * inv_m, f2.w * inv_m);
        A.u[6] = pk2(f3.x * inv_m, f3.y * inv_m); A.u[7] = pk2(f3.z * inv_m, f3.w * inv_m);
        #pragma unroll
        for (int t = 0; t < 4; ++t) {
            const unsigned short* wr = Wbf + (size_t)(16 * t + lrow) * D_ + co;
            Bv.q[0] = *(const uint4*)(wr);
            Bv.q[1] = *(const uint4*)(wr + 16);
            c[t] = __builtin_amdgcn_wmma_f32_16x16x32_bf16(false, A.v, false, Bv.v,
                                                           (short)0, c[t], false, false);
        }
    }

    // softmax over 64 cols; element r of tile t: row m0 + r + 8*hf, col 16t + (lane&15)
    #pragma unroll
    for (int r = 0; r < 8; ++r) {
        float mx = fmaxf(fmaxf(c[0][r], c[1][r]), fmaxf(c[2][r], c[3][r]));
        mx = half_red_max(mx);
        float e0 = __expf(c[0][r] - mx), e1 = __expf(c[1][r] - mx);
        float e2 = __expf(c[2][r] - mx), e3 = __expf(c[3][r] - mx);
        float sum = half_red_add(e0 + e1 + e2 + e3);
        float is = 1.f / sum;
        c[0][r] = e0 * is; c[1][r] = e1 * is; c[2][r] = e2 * is; c[3][r] = e3 * is;
    }

    // store transposed assignments (8 consecutive m per lane -> one b128) + colsum atomics
    #pragma unroll
    for (int t = 0; t < 4; ++t) {
        uint4 p;
        p.x = pk2(c[t][0], c[t][1]); p.y = pk2(c[t][2], c[t][3]);
        p.z = pk2(c[t][4], c[t][5]); p.w = pk2(c[t][6], c[t][7]);
        int col = 16 * t + lrow;
        *(uint4*)(at + ((size_t)(b * K_ + col)) * M_ + m0 + 8 * hf) = p;
        float cs = c[t][0] + c[t][1] + c[t][2] + c[t][3] + c[t][4] + c[t][5] + c[t][6] + c[t][7];
        cs += swzf<0x401f>(cs);                          // add other half (rows +-8)
        if (lane < 16) atomicAdd(&s[b * K_ + 16 * t + lane], cs);
    }
}

// ---------------- GEMM2: y[b] = A^T X, per-wave 32(k) x 32(d), m-loop ----------------
__global__ void k_gemm2(const unsigned short* __restrict__ at,
                        const unsigned short* __restrict__ xnt, float* __restrict__ y) {
    int lane = threadIdx.x & 31, w = threadIdx.x >> 5;
    int gw = blockIdx.x * 4 + w;                        // 2048 waves
    int b = gw >> 6, r6 = gw & 63;
    int k0 = (r6 >> 5) * 32, d0 = (r6 & 31) * 32;
    int lrow = lane & 15, hf = lane >> 4;

    v8f c[2][2] = {};
    union FV { v16bf v; uint4 q[2]; } Af[2], Bf[2];
    const unsigned short* ab[2]; const unsigned short* bb[2];
    #pragma unroll
    for (int i = 0; i < 2; ++i) ab[i] = at + ((size_t)(b * K_ + k0 + 16 * i + lrow)) * M_;
    #pragma unroll
    for (int j = 0; j < 2; ++j) bb[j] = xnt + ((size_t)b * D_ + d0 + 16 * j + lrow) * M_;

    for (int m0 = 0; m0 < M_; m0 += 32) {
        int off = m0 + hf * 8;
        #pragma unroll
        for (int i = 0; i < 2; ++i) {
            Af[i].q[0] = *(const uint4*)(ab[i] + off);
            Af[i].q[1] = *(const uint4*)(ab[i] + off + 16);
        }
        #pragma unroll
        for (int j = 0; j < 2; ++j) {
            Bf[j].q[0] = *(const uint4*)(bb[j] + off);
            Bf[j].q[1] = *(const uint4*)(bb[j] + off + 16);
        }
        #pragma unroll
        for (int i = 0; i < 2; ++i)
            #pragma unroll
            for (int j = 0; j < 2; ++j)
                c[i][j] = __builtin_amdgcn_wmma_f32_16x16x32_bf16(false, Af[i].v, false, Bf[j].v,
                                                                  (short)0, c[i][j], false, false);
    }
    #pragma unroll
    for (int i = 0; i < 2; ++i)
        #pragma unroll
        for (int j = 0; j < 2; ++j)
            #pragma unroll
            for (int r = 0; r < 8; ++r) {
                int kk = k0 + 16 * i + r + 8 * hf;
                int dd = d0 + 16 * j + lrow;
                y[((size_t)(b * K_ + kk)) * D_ + dd] = c[i][j][r];
            }
}

// ---------------- finalize pass 1: centroid subtract + intra-norm ----------------
__global__ void k_finalize1(const float* __restrict__ y, const float* __restrict__ s,
                            const float* __restrict__ cent, float* __restrict__ out,
                            float* __restrict__ ssb) {
    int lane = threadIdx.x & 31, w = threadIdx.x >> 5;
    int gw = blockIdx.x * 8 + w;                        // 2048 rows (b,k)
    int b = gw >> 6, k = gw & 63;
    float sk = s[b * K_ + k];
    const float* yr = y + ((size_t)(b * K_ + k)) * D_;
    const float* cr = cent + (size_t)k * D_;
    float4 v[8];
    float ss = 0.f;
    #pragma unroll
    for (int i = 0; i < 8; ++i) {
        int d = i * 128 + lane * 4;
        float4 yv = *(const float4*)(yr + d);
        float4 cv = *(const float4*)(cr + d);
        float4 t;
        t.x = yv.x - sk * cv.x; t.y = yv.y - sk * cv.y;
        t.z = yv.z - sk * cv.z; t.w = yv.w - sk * cv.w;
        v[i] = t;
        ss += t.x * t.x + t.y * t.y + t.z * t.z + t.w * t.w;
    }
    ss = wave_red_add(ss);
    float inv = 1.f / fmaxf(sqrtf(ss), 1e-12f);
    float* orow = out + ((size_t)(b * K_ + k)) * D_;
    #pragma unroll
    for (int i = 0; i < 8; ++i) {
        int d = i * 128 + lane * 4;
        float4 t = v[i];
        t.x *= inv; t.y *= inv; t.z *= inv; t.w *= inv;
        *(float4*)(orow + d) = t;
    }
    if (lane == 0) atomicAdd(&ssb[b], ss * inv * inv);
}

// ---------------- finalize pass 2: global L2 scale per batch ----------------
__global__ void k_finalize2(float* __restrict__ out, const float* __restrict__ ssb) {
    int i = blockIdx.x * 256 + threadIdx.x;             // B_*K_*D_ = 2097152
    float sc = 1.f / fmaxf(sqrtf(ssb[i >> 16]), 1e-12f);
    out[i] *= sc;
}

// ---------------- launch ----------------
extern "C" void kernel_launch(void* const* d_in, const int* in_sizes, int n_in,
                              void* d_out, int out_size, void* d_ws, size_t ws_size,
                              hipStream_t stream) {
    (void)in_sizes; (void)n_in; (void)out_size; (void)ws_size;
    const float* x    = (const float*)d_in[0];
    const float* W    = (const float*)d_in[1];
    const float* cent = (const float*)d_in[2];
    float* out = (float*)d_out;

    char* ws = (char*)d_ws;
    size_t off = 0;
    unsigned short* xnt = (unsigned short*)(ws + off); off += (size_t)B_ * D_ * M_ * 2;  // 128 MB
    unsigned short* at  = (unsigned short*)(ws + off); off += (size_t)B_ * K_ * M_ * 2;  //   8 MB
    float* y    = (float*)(ws + off); off += (size_t)B_ * K_ * D_ * 4;                    //   8 MB
    float* invn = (float*)(ws + off); off += (size_t)B_ * M_ * 4;                         // 256 KB
    unsigned short* Wbf = (unsigned short*)(ws + off); off += (size_t)K_ * D_ * 2;        // 128 KB
    float* s    = (float*)(ws + off); off += (size_t)B_ * K_ * 4;                         //   8 KB
    float* ssb  = (float*)(ws + off); off += (size_t)B_ * 4;

    k_zero<<<8, 256, 0, stream>>>(s, ssb);
    k_rownorm<<<(B_ * M_) / 8, 256, 0, stream>>>(x, invn);
    k_wconv<<<(K_ * D_) / 256, 256, 0, stream>>>(W, Wbf);
    k_transpose<<<dim3(D_ / 32, M_ / 32, B_), dim3(32, 8), 0, stream>>>(x, invn, xnt);
    k_logits_softmax<<<(B_ * M_ / 16) / 4, 128, 0, stream>>>(x, invn, Wbf, at, s);
    k_gemm2<<<(B_ * 2 * 32) / 4, 128, 0, stream>>>(at, xnt, y);
    k_finalize1<<<(B_ * K_) / 8, 256, 0, stream>>>(y, s, cent, out, ssb);
    k_finalize2<<<(B_ * K_ * D_) / 256, 256, 0, stream>>>(out, ssb);
}
